// Channel_dot_80951543595325
// MI455X (gfx1250) — matrix operand order repaired and checked by reading, not compiled, not run on hardware
//
#include <hip/hip_runtime.h>
#include <hip/hip_bf16.h>

// Problem constants (match reference)
#define B_    4
#define T_    5
#define C_    512
#define HW_   1024
#define THW_  5120
#define F_    16
#define G3_   16384

// g-GEMM staging geometry
#define KSLAB 64                  // K elements staged per buffer
#define BLD   72                  // padded LDS row stride in bf16 (144 B -> conflict-free)
#define BUFBYTES (64 * BLD * 2)   // 9216 B per buffer

#if __has_builtin(__builtin_amdgcn_tensor_load_to_lds)
#define HAVE_TDM 1
#else
#define HAVE_TDM 0
#endif

typedef __attribute__((ext_vector_type(16))) __bf16 v16bf;
typedef __attribute__((ext_vector_type(8)))  __bf16 v8bf;
typedef __attribute__((ext_vector_type(8)))  float  v8f;
typedef __attribute__((ext_vector_type(4)))  unsigned int v4u;
typedef __attribute__((ext_vector_type(8)))  int v8i;
typedef __attribute__((ext_vector_type(4)))  int v4i;

// ---------------------------------------------------------------------------
// WMMA helpers (CDNA5, wave32). D = A(16x32 bf16) * B(32x16 bf16) + C(16x16 f32)
// ---------------------------------------------------------------------------
__device__ __forceinline__ v8f wmma_bf16(v16bf a, v16bf b, v8f c) {
  return __builtin_amdgcn_wmma_f32_16x16x32_bf16(
      /*neg_a=*/false, a, /*neg_b=*/false, b,
      /*c_mod=*/(short)0, c, /*reuse_a=*/false, /*reuse_b=*/false);
}

// Opaque barrier: tells LLVM the LDS buffer may have been written externally
// (by the TDM engine). Without this, LLVM sees no store to the __shared__
// array and folds every ds_load to undef (observed: ds=0 in round 3).
__device__ __forceinline__ void lds_published(void* p) {
  asm volatile("" : "+v"(p) :: "memory");
}

// A fragment 16x32 from row-major f32 [M][K] with on-the-fly bf16 convert.
__device__ __forceinline__ v16bf load_a_f32cvt(const float* __restrict__ A, int lda,
                                               int m0, int k0, int lane) {
  int row = m0 + (lane & 15);
  int h   = (lane >> 4) & 1;
  const float* p = A + (size_t)row * lda + k0 + 16 * h;
  v8f c0 = *(const v8f*)p;
  v8f c1 = *(const v8f*)(p + 8);
  v16bf a;
#pragma unroll
  for (int j = 0; j < 8; ++j) { a[j] = (__bf16)c0[j]; a[8 + j] = (__bf16)c1[j]; }
  return a;
}

// A fragment 16x32 from row-major bf16 [M][K].
__device__ __forceinline__ v16bf load_a_bf16(const __bf16* __restrict__ A, int lda,
                                             int m0, int k0, int lane) {
  int row = m0 + (lane & 15);
  int h   = (lane >> 4) & 1;
  const __bf16* p = A + (size_t)row * lda + k0 + 16 * h;
  v8bf c0 = *(const v8bf*)p;
  v8bf c1 = *(const v8bf*)(p + 8);
  v16bf a;
#pragma unroll
  for (int j = 0; j < 8; ++j) { a[j] = c0[j]; a[8 + j] = c1[j]; }
  return a;
}

// B fragment 32x16 from [N][K] row-major bf16 in GLOBAL memory.
__device__ __forceinline__ v16bf load_b_nk(const __bf16* __restrict__ Bm, int ldk,
                                           int n0, int k0, int lane) {
  int col = n0 + (lane & 15);
  int h   = (lane >> 4) & 1;
  return *(const v16bf*)(Bm + (size_t)col * ldk + k0 + 16 * h);
}

// B fragment 32x16 from LDS tile [64][BLD] bf16 (two ds_load_b128, 16B aligned).
__device__ __forceinline__ v16bf load_b_lds(const __bf16* Bs, int n0, int k0, int lane) {
  int row = n0 + (lane & 15);
  int h   = (lane >> 4) & 1;
  const __bf16* p = Bs + row * BLD + k0 + 16 * h;
  v8bf lo = *(const v8bf*)p;
  v8bf hi = *(const v8bf*)(p + 8);
  v16bf r;
#pragma unroll
  for (int j = 0; j < 8; ++j) { r[j] = lo[j]; r[8 + j] = hi[j]; }
  return r;
}

#if HAVE_TDM
// Issue a TDM DMA of one 2D tile (KSLAB k x 64 c rows, bf16) from x1t[b] into
// LDS at byte offset ldsoff. D# built per CDNA5 ISA.
// Padding: 4 DWORDs every 32 DWORDs (one 128B row) -> LDS row stride 144 B.
// 6-arg builtin form: (uint32x4 g0, int32x8 g1, int32x4, int32x4, int32x8, i32 cpol)
__device__ __forceinline__ void tdm_issue(const __bf16* gsrc, unsigned ldsoff) {
  unsigned long long ga = (unsigned long long)(size_t)gsrc;
  v4u g0;
  g0[0] = 1u;                                              // count=1 (valid user D#)
  g0[1] = ldsoff;                                          // lds_addr (bytes)
  g0[2] = (unsigned)ga;                                    // global_addr[31:0]
  g0[3] = (unsigned)((ga >> 32) & 0x1FFFFFFu) | (2u << 30);// global_addr[56:32] | type=2
  v8i g1;
  g1[0] = (int)((1u << 16)      // data_size = 1 -> 2 bytes
              | (1u << 20)      // pad_enable
              | (4u << 22)      // pad_interval: 2^(4+1)=32 DWORDs (=128B row)
              | (3u << 25));    // pad_amount: 4 DWORDs (16B)
  g1[1] = (int)((THW_ & 0xFFFF) << 16);                           // tensor_dim0[15:0]
  g1[2] = (int)(((THW_ >> 16) & 0xFFFF) | ((C_ & 0xFFFF) << 16)); // td0[31:16]|td1[15:0]
  g1[3] = (int)(((C_ >> 16) & 0xFFFF) | ((unsigned)KSLAB << 16)); // td1[31:16]|tile_dim0
  g1[4] = (int)(64u);                                             // tile_dim1=64, tile_dim2=0
  g1[5] = (int)THW_;                                              // tensor_dim0_stride[31:0]
  g1[6] = 0;                                                      // stride hi | td1_stride lo
  g1[7] = 0;
  v4i z4 = {0, 0, 0, 0};                                          // groups 2/3 unused (2D)
  v8i z8 = {0, 0, 0, 0, 0, 0, 0, 0};
  __builtin_amdgcn_tensor_load_to_lds(g0, g1, z4, z4, z8, 0);
}
#else
// Fallback: cooperative stage of the same tile (adds ds_store path).
__device__ __forceinline__ void stage_tile(const __bf16* __restrict__ Xrow0, int kbase,
                                           __bf16* dst) {
  int t   = threadIdx.x;        // 256 threads, 32B each
  int row = t >> 2;             // 0..63
  int seg = (t & 3) * 16;       // 0,16,32,48
  const __bf16* src = Xrow0 + (size_t)row * THW_ + kbase + seg;
  v8bf a0 = *(const v8bf*)src;
  v8bf a1 = *(const v8bf*)(src + 8);
  *(v8bf*)(dst + row * BLD + seg)     = a0;
  *(v8bf*)(dst + row * BLD + seg + 8) = a1;
}
#endif

// ---------------------------------------------------------------------------
// 1) Reshape + convert: input[b,t,c,h,w] f32 -> x_t[b][c][t*HW+hw] bf16
// ---------------------------------------------------------------------------
__global__ void reshape_cvt_kernel(const float* __restrict__ in, __bf16* __restrict__ out) {
  size_t idx = (size_t)blockIdx.x * blockDim.x + threadIdx.x;
  int    hw  = (int)(idx % HW_);
  size_t r   = idx / HW_;
  int    c   = (int)(r % C_);  r /= C_;
  int    t   = (int)(r % T_);
  int    b   = (int)(r / T_);
  out[((size_t)(b * C_ + c)) * THW_ + (size_t)t * HW_ + hw] = (__bf16)in[idx];
}

// ---------------------------------------------------------------------------
// 2) q/k GEMM: q[f,c] = sum_i W[f,i] * x_t[c,i] + bias[f]; out q_t[b][c][f] (K pad 32)
// ---------------------------------------------------------------------------
__global__ void qk_kernel(const float* __restrict__ Wm, const float* __restrict__ bias,
                          const __bf16* __restrict__ X, __bf16* __restrict__ Q) {
  int lane = threadIdx.x & 31;
  int wv   = threadIdx.x >> 5;
  int b    = blockIdx.y;
  int c0   = blockIdx.x * 64 + wv * 16;
  const __bf16* Xb = X + (size_t)b * C_ * THW_;
  v8f acc = {};
  for (int k = 0; k < THW_; k += 32) {
    v16bf a  = load_a_f32cvt(Wm, THW_, 0, k, lane);
    v16bf bb = load_b_nk(Xb, THW_, c0, k, lane);
    acc = wmma_bf16(a, bb, acc);
  }
  int h = lane >> 4;
  int c = c0 + (lane & 15);
  __bf16* qrow = Q + ((size_t)(b * C_) + c) * 32;
  v8bf outv, z;
#pragma unroll
  for (int v = 0; v < 8; ++v) {
    outv[v] = (__bf16)(acc[v] + bias[8 * h + v]);   // f = v + 8h
    z[v]    = (__bf16)0.0f;
  }
  *(v8bf*)(qrow + 8 * h)      = outv;
  *(v8bf*)(qrow + 16 + 8 * h) = z;
}

// ---------------------------------------------------------------------------
// 3) scores[b,c,d] = sum_f q_t[c,f] * k_t[d,f]  (K=32 padded, one WMMA/tile)
// ---------------------------------------------------------------------------
__global__ void scores_kernel(const __bf16* __restrict__ Qt, const __bf16* __restrict__ Kt,
                              float* __restrict__ Sc) {
  int lane = threadIdx.x & 31;
  int wv   = threadIdx.x >> 5;
  int b    = blockIdx.z;
  int c0   = (blockIdx.y * 8 + wv) * 16;
  int d0   = blockIdx.x * 16;
  const __bf16* Qb = Qt + (size_t)b * C_ * 32;
  const __bf16* Kb = Kt + (size_t)b * C_ * 32;
  v16bf a  = load_a_bf16(Qb, 32, c0, 0, lane);
  v16bf bb = load_b_nk(Kb, 32, d0, 0, lane);
  v8f acc = {};
  acc = wmma_bf16(a, bb, acc);
  int h = lane >> 4;
  int d = d0 + (lane & 15);
#pragma unroll
  for (int v = 0; v < 8; ++v) {
    int c = c0 + v + 8 * h;
    Sc[((size_t)(b * C_) + c) * C_ + d] = acc[v];
  }
}

// ---------------------------------------------------------------------------
// 4) Softmax over axis c per (b,d) column; write bf16 S[b][c][d].
// ---------------------------------------------------------------------------
__global__ void softmax_kernel(const float* __restrict__ Sc, __bf16* __restrict__ Sb) {
  int b = blockIdx.y;
  int d = blockIdx.x * blockDim.x + threadIdx.x;
  const float* col = Sc + (size_t)b * C_ * C_ + d;
  float m = -3.4e38f;
  for (int c = 0; c < C_; ++c) m = fmaxf(m, col[(size_t)c * C_]);
  float s = 0.f;
  for (int c = 0; c < C_; ++c) s += __expf(col[(size_t)c * C_] - m);
  float inv = 1.f / s;
  __bf16* ocol = Sb + (size_t)b * C_ * C_ + d;
  for (int c = 0; c < C_; ++c)
    ocol[(size_t)c * C_] = (__bf16)(__expf(col[(size_t)c * C_] - m) * inv);
}

// ---------------------------------------------------------------------------
// 5) g GEMM (dominant): g[b,o,c] = sum_i W3[o,i]*x1t[c,i] + b3[o], stored [b][o][c] bf16.
//    Grid (8 c-macro, 4 batch, 128 o-macro): the 32 blocks sharing one W3
//    row-strip are consecutive in dispatch -> L2 captures reuse (~320MB HBM).
//    B tile double-buffered in LDS via TDM (tensor_load_to_lds + s_wait_tensorcnt),
//    issued by wave 0; all 8 waves consume via ds_load_b128 (conflict-free, BLD=72).
// ---------------------------------------------------------------------------
__global__ void g_gemm_kernel(const float* __restrict__ W3, const float* __restrict__ b3,
                              const __bf16* __restrict__ X1, __bf16* __restrict__ G) {
  __shared__ __bf16 Bs[2][64 * BLD];
  int lane = threadIdx.x & 31;
  int wv   = threadIdx.x >> 5;                 // 0..7
  int b    = blockIdx.y;
  int c0   = blockIdx.x * 64;
  int o0   = blockIdx.z * 128 + wv * 16;
  const __bf16* Xb = X1 + (size_t)b * C_ * THW_ + (size_t)c0 * THW_;  // row c0, k=0
  v8f acc[4] = {};
  const float* arow = W3 + (size_t)(o0 + (lane & 15)) * THW_;
  const int NS = THW_ / KSLAB;                 // 80 stages

#if HAVE_TDM
  if (threadIdx.x < 32) tdm_issue(Xb, 0u);     // prologue: stage 0 -> buf0
#else
  stage_tile(Xb, 0, &Bs[0][0]);
#endif

  for (int s = 0; s < NS; ++s) {
    int p = s & 1;
#if HAVE_TDM
    if (threadIdx.x < 32) {
      if (s + 1 < NS) {
        tdm_issue(Xb + (size_t)(s + 1) * KSLAB, (unsigned)((p ^ 1) * BUFBYTES));
        __builtin_amdgcn_s_wait_tensorcnt(1);  // buf[p] landed (next DMA in flight)
      } else {
        __builtin_amdgcn_s_wait_tensorcnt(0);  // last stage: drain everything
      }
    }
#endif
    __syncthreads();                           // publish buf[p]
    lds_published(&Bs[p][0]);                  // TDM wrote here: keep the ds_loads
    int kbase = s * KSLAB;
    if (kbase + KSLAB < THW_) __builtin_prefetch(arow + kbase + KSLAB, 0, 1);
#pragma unroll
    for (int kk = 0; kk < KSLAB; kk += 32) {
      v16bf a = load_a_f32cvt(W3, THW_, o0, kbase + kk, lane);
#pragma unroll
      for (int n = 0; n < 4; ++n) {
        v16bf bb = load_b_lds(&Bs[p][0], 16 * n, kk, lane);
        acc[n] = wmma_bf16(a, bb, acc[n]);
      }
    }
    __syncthreads();                           // release buf[p] before re-targeting
#if !HAVE_TDM
    if (s + 1 < NS) stage_tile(Xb, (s + 1) * KSLAB, &Bs[p ^ 1][0]);
#endif
  }

  int h  = lane >> 4;
  int cc = lane & 15;
#pragma unroll
  for (int n = 0; n < 4; ++n) {
    int c = c0 + 16 * n + cc;
#pragma unroll
    for (int v = 0; v < 8; ++v) {
      int o = o0 + v + 8 * h;
      G[((size_t)b * G3_ + o) * C_ + c] = (__bf16)(acc[n][v] + b3[o]);
    }
  }
}

// ---------------------------------------------------------------------------
// 6) Final GEMM + fused output permute:
//    out[b,c,o] = sum_d S[c,d]*g[o,d];  d_out[b,f,c,h,w] = out[b,c,f*HW+hw]
//    Grid (8 c-macro, 4 batch, 128 o-macro) for the same L2-locality reason.
// ---------------------------------------------------------------------------
__global__ void out_gemm_kernel(const __bf16* __restrict__ Sb, const __bf16* __restrict__ G,
                                float* __restrict__ Out) {
  int lane = threadIdx.x & 31;
  int wv   = threadIdx.x >> 5;                 // 0..7
  int b    = blockIdx.y;
  int c0   = blockIdx.x * 64 + (wv >> 1) * 16;
  int o0   = blockIdx.z * 128 + (wv & 1) * 64;
  const __bf16* Sbb = Sb + (size_t)b * C_ * C_;
  const __bf16* Gb  = G  + (size_t)b * G3_ * C_;
  v8f acc[4] = {};
  for (int k = 0; k < C_; k += 32) {
    v16bf a = load_a_bf16(Sbb, C_, c0, k, lane);
#pragma unroll
    for (int n = 0; n < 4; ++n) {
      v16bf bb = load_b_nk(Gb, C_, o0 + 16 * n, k, lane);
      acc[n] = wmma_bf16(a, bb, acc[n]);
    }
  }
  int h  = lane >> 4;
  int nc = lane & 15;
#pragma unroll
  for (int n = 0; n < 4; ++n) {
    int o  = o0 + 16 * n + nc;
    int f  = o >> 10;
    int hw = o & 1023;
#pragma unroll
    for (int v = 0; v < 8; ++v) {
      int c = c0 + v + 8 * h;
      Out[(((size_t)b * F_ + f) * C_ + c) * HW_ + hw] = acc[n][v];
    }
  }
}

// ---------------------------------------------------------------------------
// Host launch
// ---------------------------------------------------------------------------
extern "C" void kernel_launch(void* const* d_in, const int* in_sizes, int n_in,
                              void* d_out, int out_size, void* d_ws, size_t ws_size,
                              hipStream_t stream) {
  const float* input1 = (const float*)d_in[0];
  const float* input2 = (const float*)d_in[1];
  const float* W1     = (const float*)d_in[2];
  const float* b1     = (const float*)d_in[3];
  const float* W2     = (const float*)d_in[4];
  const float* b2     = (const float*)d_in[5];
  const float* W3     = (const float*)d_in[6];
  const float* b3     = (const float*)d_in[7];
  float* out = (float*)d_out;

  // Workspace layout (bytes), all 256-aligned (~115.6 MB total)
  char* ws = (char*)d_ws;
  __bf16* x1t = (__bf16*)(ws + 0);            // [4][512][5120] bf16
  __bf16* x2t = (__bf16*)(ws + 20971520);     // [4][512][5120] bf16
  __bf16* qt  = (__bf16*)(ws + 41943040);     // [4][512][32]   bf16 (K padded)
  __bf16* kt  = (__bf16*)(ws + 42074112);     // [4][512][32]   bf16
  float*  sc  = (float* )(ws + 42205184);     // [4][512][512]  f32
  __bf16* sb  = (__bf16*)(ws + 46399488);     // [4][512][512]  bf16
  __bf16* g   = (__bf16*)(ws + 48496640);     // [4][16384][512] bf16

  reshape_cvt_kernel<<<40960, 256, 0, stream>>>(input1, x1t);
  reshape_cvt_kernel<<<40960, 256, 0, stream>>>(input2, x2t);
  qk_kernel<<<dim3(8, 4), 128, 0, stream>>>(W1, b1, x1t, qt);
  qk_kernel<<<dim3(8, 4), 128, 0, stream>>>(W2, b2, x2t, kt);
  scores_kernel<<<dim3(32, 4, 4), 256, 0, stream>>>(qt, kt, sc);
  softmax_kernel<<<dim3(2, 4), 256, 0, stream>>>(sc, sb);
  g_gemm_kernel<<<dim3(8, 4, 128), 256, 0, stream>>>(W3, b3, x1t, g);
  out_gemm_kernel<<<dim3(8, 4, 128), 256, 0, stream>>>(sb, g, out);
}
